// NemotronHMOE_13580686590238
// MI455X (gfx1250) — compile-verified, matrix operand order checked by reading, not run on hardware
//
#include <hip/hip_runtime.h>

#define NUM_TOKENS  8192
#define HIDDEN      7168
#define NUM_EXPERTS 256
#define TOP_K       8
#define N_GROUP     8
#define TOPK_GROUP  4
#define ROUTED_SCALE 2.5f

typedef __attribute__((ext_vector_type(16))) __bf16        v16bf;
typedef __attribute__((ext_vector_type(2)))  __bf16        v2bf;
typedef __attribute__((ext_vector_type(8)))  float         v8f;
typedef __attribute__((ext_vector_type(8)))  unsigned int  v8u;

// Pack two fp32 -> packed bf16x2. Let the compiler lower the fptrunc pair;
// on gfx1250 this should select the native packed convert (v_cvt_pk_bf16_f32).
__device__ __forceinline__ unsigned int pk_bf16(float lo, float hi) {
  v2bf r;
  r[0] = (__bf16)lo;
  r[1] = (__bf16)hi;
  return __builtin_bit_cast(unsigned int, r);
}

// Build one lane's v16bf operand from 16 fp32 values (two 8-K runs, 16 apart) laid out
// per the CDNA5 16-bit A/B WMMA layout (two K-octets per lane half).
__device__ __forceinline__ v16bf pack16(float4 a, float4 b, float4 c, float4 d) {
  v8u t;
  t[0] = pk_bf16(a.x, a.y); t[1] = pk_bf16(a.z, a.w);
  t[2] = pk_bf16(b.x, b.y); t[3] = pk_bf16(b.z, b.w);
  t[4] = pk_bf16(c.x, c.y); t[5] = pk_bf16(c.z, c.w);
  t[6] = pk_bf16(d.x, d.y); t[7] = pk_bf16(d.z, d.w);
  return __builtin_bit_cast(v16bf, t);
}

__device__ __forceinline__ v16bf load_pack_row(const float* __restrict__ row, int k, int kh8) {
  const float* p = row + k + kh8;
  float4 a = *(const float4*)(p);
  float4 b = *(const float4*)(p + 4);
  float4 c = *(const float4*)(p + 16);
  float4 d = *(const float4*)(p + 20);
  return pack16(a, b, c, d);
}

// ---------------------------------------------------------------------------
// Kernel 1: router GEMM.  logits[T,E] = hidden[T,H] * weight[E,H]^T  (fp32 out)
// Block: 256 threads = 8 waves arranged 2(M) x 4(N). Block tile: M=32, N=256.
// Each wave: one 16-row A slice, four 16x16 N-tiles, K-step = 32 (bf16 WMMA).
// ---------------------------------------------------------------------------
__global__ __launch_bounds__(256) void dsv3_router_gemm(
    const float* __restrict__ hs, const float* __restrict__ wt,
    float* __restrict__ logits) {
  const int lane = threadIdx.x & 31;
  const int wave = threadIdx.x >> 5;
  const int wm   = wave >> 2;      // 0..1  (M sub-tile)
  const int wn   = wave & 3;       // 0..3  (N quarter)
  const int l16  = lane & 15;
  const int half = lane >> 4;      // lane half selects K-octet
  const int kh8  = half * 8;

  const int m0 = blockIdx.x * 32 + wm * 16;
  const int n0 = wn * 64;

  const float* arow  = hs + (size_t)(m0 + l16) * HIDDEN;
  const float* brow0 = wt + (size_t)(n0 +  0 + l16) * HIDDEN;
  const float* brow1 = wt + (size_t)(n0 + 16 + l16) * HIDDEN;
  const float* brow2 = wt + (size_t)(n0 + 32 + l16) * HIDDEN;
  const float* brow3 = wt + (size_t)(n0 + 48 + l16) * HIDDEN;

  v8f c0 = {}, c1 = {}, c2 = {}, c3 = {};

  for (int k = 0; k < HIDDEN; k += 32) {
    v16bf a  = load_pack_row(arow,  k, kh8);
    v16bf b0 = load_pack_row(brow0, k, kh8);
    v16bf b1 = load_pack_row(brow1, k, kh8);
    v16bf b2 = load_pack_row(brow2, k, kh8);
    v16bf b3 = load_pack_row(brow3, k, kh8);
    c0 = __builtin_amdgcn_wmma_f32_16x16x32_bf16(false, a, false, b0, (short)0, c0, false, false);
    c1 = __builtin_amdgcn_wmma_f32_16x16x32_bf16(false, a, false, b1, (short)0, c1, false, false);
    c2 = __builtin_amdgcn_wmma_f32_16x16x32_bf16(false, a, false, b2, (short)0, c2, false, false);
    c3 = __builtin_amdgcn_wmma_f32_16x16x32_bf16(false, a, false, b3, (short)0, c3, false, false);
  }

  // C layout: lanes 0-15 -> N=lane, M=vgpr r; lanes 16-31 -> N=lane-16, M=8+r.
  float* crow = logits + (size_t)(m0 + half * 8) * NUM_EXPERTS + l16;
#pragma unroll
  for (int r = 0; r < 8; ++r) {
    float* cr = crow + (size_t)r * NUM_EXPERTS;
    cr[n0 +  0] = c0[r];
    cr[n0 + 16] = c1[r];
    cr[n0 + 32] = c2[r];
    cr[n0 + 48] = c3[r];
  }
}

// ---------------------------------------------------------------------------
// Kernel 2: sigmoid + bias, group top-2 sums, top-4 groups, top-8 experts,
// normalize, emit sorted (indices, values). One wave per token.
// ---------------------------------------------------------------------------
__global__ __launch_bounds__(256) void dsv3_router_topk(
    const float* __restrict__ logits, const float* __restrict__ bias,
    int* __restrict__ outIdx, float* __restrict__ outVal) {
  const int lane = threadIdx.x & 31;
  const int wave = threadIdx.x >> 5;
  const int tok  = blockIdx.x * 8 + wave;

  const float* lrow = logits + (size_t)tok * NUM_EXPERTS + lane * 8;
  float4 x0 = *(const float4*)(lrow);
  float4 x1 = *(const float4*)(lrow + 4);
  float4 bb0 = *(const float4*)(bias + lane * 8);
  float4 bb1 = *(const float4*)(bias + lane * 8 + 4);

  float xv[8]  = {x0.x, x0.y, x0.z, x0.w, x1.x, x1.y, x1.z, x1.w};
  float bv[8]  = {bb0.x, bb0.y, bb0.z, bb0.w, bb1.x, bb1.y, bb1.z, bb1.w};
  float s[8], swb[8];
#pragma unroll
  for (int i = 0; i < 8; ++i) {
    s[i]   = 1.0f / (1.0f + __expf(-xv[i]));   // sigmoid
    swb[i] = s[i] + bv[i];
  }

  // per-lane top-2 of biased scores (8 experts/lane; 4 lanes = 1 group of 32)
  float t1 = -__builtin_inff(), t2 = -__builtin_inff();
#pragma unroll
  for (int i = 0; i < 8; ++i) {
    if (swb[i] > t1)      { t2 = t1; t1 = swb[i]; }
    else if (swb[i] > t2) { t2 = swb[i]; }
  }
  // merge top-2 across the 4 lanes of the group
#pragma unroll
  for (int off = 1; off <= 2; off <<= 1) {
    float o1 = __shfl_xor(t1, off, 32);
    float o2 = __shfl_xor(t2, off, 32);
    float m1 = fmaxf(t1, o1);
    float m2 = fmaxf(fminf(t1, o1), (t1 >= o1) ? t2 : o2);
    t1 = m1; t2 = m2;
  }
  float gscore = t1 + t2;                 // valid on all lanes of the group
  const int myg = lane >> 2;

  // broadcast all 8 group scores; select top-4 groups by rank (ties: low idx)
  float gsv[8];
#pragma unroll
  for (int g = 0; g < 8; ++g) gsv[g] = __shfl(gscore, g * 4, 32);
  int rank = 0;
#pragma unroll
  for (int g = 0; g < 8; ++g)
    rank += (gsv[g] > gscore) || (gsv[g] == gscore && g < myg);
  const bool keep = (rank < TOPK_GROUP);

  float mval[8];
#pragma unroll
  for (int i = 0; i < 8; ++i) mval[i] = keep ? swb[i] : 0.0f;

  // iterative wave-wide top-8 of masked biased scores (ties -> lower index)
  int   selI[8]; float selS[8];
#pragma unroll
  for (int it = 0; it < TOP_K; ++it) {
    float bvv = mval[0]; int bi = 0; float bs = s[0];
#pragma unroll
    for (int i = 1; i < 8; ++i)
      if (mval[i] > bvv) { bvv = mval[i]; bi = i; bs = s[i]; }
    int bidx = lane * 8 + bi;
#pragma unroll
    for (int off = 16; off >= 1; off >>= 1) {
      float ov = __shfl_xor(bvv, off, 32);
      int   oi = __shfl_xor(bidx, off, 32);
      float os = __shfl_xor(bs, off, 32);
      if (ov > bvv || (ov == bvv && oi < bidx)) { bvv = ov; bidx = oi; bs = os; }
    }
    selI[it] = bidx; selS[it] = bs;
    if (lane == (bidx >> 3)) {
#pragma unroll
      for (int i = 0; i < 8; ++i)
        if (i == (bidx & 7)) mval[i] = -__builtin_inff();
    }
  }

  float sum = 0.0f;
#pragma unroll
  for (int it = 0; it < TOP_K; ++it) sum += selS[it];
  const float scale = ROUTED_SCALE / (sum + 1e-20f);
  float val[8];
#pragma unroll
  for (int it = 0; it < TOP_K; ++it) val[it] = selS[it] * scale;

  // final ordering = top_k over normalized scores: desc value, asc index ties
#pragma unroll
  for (int a = 0; a < 8; ++a) {
#pragma unroll
    for (int b = 0; b < 7; ++b) {
      bool sw = (val[b + 1] > val[b]) ||
                (val[b + 1] == val[b] && selI[b + 1] < selI[b]);
      if (sw) {
        float tv = val[b]; val[b] = val[b + 1]; val[b + 1] = tv;
        int   ti = selI[b]; selI[b] = selI[b + 1]; selI[b + 1] = ti;
      }
    }
  }

  if (lane == 0) {
#pragma unroll
    for (int i = 0; i < 8; ++i) outIdx[(size_t)tok * 8 + i] = selI[i];
#pragma unroll
    for (int i = 0; i < 8; ++i) outVal[(size_t)tok * 8 + i] = val[i];
  }
}

extern "C" void kernel_launch(void* const* d_in, const int* in_sizes, int n_in,
                              void* d_out, int out_size, void* d_ws, size_t ws_size,
                              hipStream_t stream) {
  const float* hs   = (const float*)d_in[0];   // [8192, 7168]
  const float* wt   = (const float*)d_in[1];   // [256, 7168]
  const float* bias = (const float*)d_in[2];   // [256]

  float* logits = (float*)d_ws;                            // 8192*256*4 = 8 MB
  int*   outIdx = (int*)d_out;                             // first 8192*8 int32
  float* outVal = (float*)d_out + (size_t)NUM_TOKENS * 8;  // then 8192*8 fp32

  dsv3_router_gemm<<<dim3(NUM_TOKENS / 32), dim3(256), 0, stream>>>(hs, wt, logits);
  dsv3_router_topk<<<dim3(NUM_TOKENS / 8), dim3(256), 0, stream>>>(logits, bias, outIdx, outVal);
}
